// MPN_51462298140813
// MI455X (gfx1250) — matrix operand order repaired and checked by reading, not compiled, not run on hardware
//
#include <hip/hip_runtime.h>
#include <hip/hip_bf16.h>

typedef __bf16 bf16_t;
typedef __attribute__((ext_vector_type(16))) __bf16 v16bf;
typedef __attribute__((ext_vector_type(8)))  __bf16 v8bf;
typedef __attribute__((ext_vector_type(4)))  __bf16 v4bf;
typedef __attribute__((ext_vector_type(8)))  float  v8f;

#define T_  512
#define B_  64
#define NX_ 512
#define NH_ 1024
#define NY_ 512
#define NC_ 32          // M-columns per block (NH_/NC_ = 32 blocks)

// LDS row strides (padded: 16B-aligned rows, bank-conflict-free lane strides)
#define MS_LD 516       // f32  rows (516*4 = 2064 B, 516%64 == 4)
#define WS_LD 516       // f32  rows
#define XS_LD 520       // bf16 rows (520*2 = 1040 B)
#define XT_LD 72        // bf16 rows (72*2  = 144 B)
#define HS_LD 72        // bf16 rows

static __device__ __forceinline__ v16bf cat8(v8bf lo, v8bf hi) {
  return __builtin_shufflevector(lo, hi, 0,1,2,3,4,5,6,7,8,9,10,11,12,13,14,15);
}

// Branch-free tanh on CDNA5 transcendental units. Prefers the native V_TANH
// trans op if the toolchain exposes it; otherwise 5 straight-line VALU ops
// (v_mul, v_exp, v_add, v_rcp, v_fma) with correct +/-1 saturation.
static __device__ __forceinline__ float fast_tanh(float x) {
#if __has_builtin(__builtin_amdgcn_tanh_f32)
  return __builtin_amdgcn_tanh_f32(x);
#elif __has_builtin(__builtin_amdgcn_tanhf)
  return __builtin_amdgcn_tanhf(x);
#else
  // tanh(x) = 1 - 2/(1 + e^{2x});  e^{2x} = exp2(x * 2*log2(e))
  float e = __builtin_amdgcn_exp2f(x * 2.88539008177792681f);
  return 1.0f - 2.0f * __builtin_amdgcn_rcpf(1.0f + e);
#endif
}

// ---------------------------------------------------------------------------
// Phase 1: per-column-slice fast-weight recurrence. One block owns NC_ columns
// of M (f32, LDS-resident, ~271 KB total LDS of the 320 KB/WGP) and loops over
// all T steps, emitting h in bf16.
// ---------------------------------------------------------------------------
__global__ __launch_bounds__(128) void mpn_recur(
    const float* __restrict__ x,      // [T,B,NX] f32
    const float* __restrict__ Wi,     // [NX,NH]  f32
    const float* __restrict__ eta_p,  // scalar
    const float* __restrict__ lam_p,  // scalar
    bf16_t* __restrict__ hout)        // [T,B,NH] bf16 (workspace)
{
  __shared__ float  Ms[NC_ * MS_LD];   // Ms[j][k] = M[k, j0g+j]
  __shared__ float  Ws[NC_ * WS_LD];   // Ws[j][k] = Wi[k, j0g+j]
  __shared__ bf16_t xs [B_  * XS_LD];  // xs[b][k]  = x_t[b,k]
  __shared__ bf16_t xTs[NX_ * XT_LD];  // xTs[k][b] = x_t[b,k]
  __shared__ bf16_t hs [NC_ * HS_LD];  // hs[j][b]  = h[b,j]

  const int tid  = threadIdx.x;        // 0..127 (4 waves)
  const int wave = tid >> 5;
  const int lane = tid & 31;
  const int half = lane >> 4;
  const int lm   = lane & 15;
  const int j0g  = blockIdx.x * NC_;   // global column base

  const float etaV = eta_p[0];
  const float lamV = lam_p[0];

  // Stage constant Wi slice (coalesced: 32 consecutive j per 32 tids) + zero M
  for (int i = tid; i < NC_ * NX_; i += 128) {
    int jj = i & (NC_ - 1);
    int k  = i >> 5;
    Ws[jj * WS_LD + k] = Wi[(size_t)k * NH_ + j0g + jj];
  }
  for (int i = tid; i < NC_ * MS_LD; i += 128) Ms[i] = 0.0f;
  __syncthreads();

  const v8f vzero = {0.f,0.f,0.f,0.f,0.f,0.f,0.f,0.f};

  for (int t = 0; t < T_; ++t) {
    // ---- stage x_t into xs (row-major) and xTs (transposed), f32 -> bf16 ---
    {
      const float4* xg  = (const float4*)(x + (size_t)t * (B_ * NX_));
      const int tn = (t + 1 < T_) ? (t + 1) : t;
      const float4* xgn = (const float4*)(x + (size_t)tn * (B_ * NX_));
      for (int i = tid; i < (B_ * NX_) / 4; i += 128) {
        if ((tid & 7) == 0) __builtin_prefetch(&xgn[i], 0, 0);  // next step
        float4 v = xg[i];
        int b = i / (NX_ / 4);
        int k = (i % (NX_ / 4)) * 4;
        v4bf xv;
        xv[0] = (bf16_t)v.x; xv[1] = (bf16_t)v.y;
        xv[2] = (bf16_t)v.z; xv[3] = (bf16_t)v.w;
        *(v4bf*)&xs[b * XS_LD + k] = xv;
        xTs[(k + 0) * XT_LD + b] = xv[0];
        xTs[(k + 1) * XT_LD + b] = xv[1];
        xTs[(k + 2) * XT_LD + b] = xv[2];
        xTs[(k + 3) * XT_LD + b] = xv[3];
      }
    }
    __syncthreads();

    // ---- forward GEMM: h[b, j] = tanh( sum_k x[b,k] * Wi[k,j]*(1+M[k,j]) ) --
    // wave w owns b-rows [16w, 16w+16), both 16-wide j-tiles.
    {
      const int brow = wave * 16;
      v8f acc[2]; acc[0] = vzero; acc[1] = vzero;

      for (int k0 = 0; k0 < NX_; k0 += 32) {
        // A fragment: 16x32 tile of x (K chunks at 16*half and 8+16*half)
        const bf16_t* xrow = &xs[(brow + lm) * XS_LD + k0];
        v16bf a = cat8(*(const v8bf*)(xrow + 16 * half),
                       *(const v8bf*)(xrow + 8 + 16 * half));
        #pragma unroll
        for (int jt = 0; jt < 2; ++jt) {
          int j = jt * 16 + lm;
          const float4* wp = (const float4*)&Ws[j * WS_LD + k0 + 16 * half];
          const float4* mp = (const float4*)&Ms[j * MS_LD + k0 + 16 * half];
          v16bf bb;
          #pragma unroll
          for (int q = 0; q < 4; ++q) {
            float4 wv = wp[q]; float4 mv = mp[q];
            bb[4*q+0] = (bf16_t)(wv.x * (1.0f + mv.x));
            bb[4*q+1] = (bf16_t)(wv.y * (1.0f + mv.y));
            bb[4*q+2] = (bf16_t)(wv.z * (1.0f + mv.z));
            bb[4*q+3] = (bf16_t)(wv.w * (1.0f + mv.w));
          }
          acc[jt] = __builtin_amdgcn_wmma_f32_16x16x32_bf16(
              false, a, false, bb, (short)0, acc[jt], false, false);
        }
      }
      // epilogue: tanh, stash h tile in LDS (transposed) + stream to global
      #pragma unroll
      for (int jt = 0; jt < 2; ++jt) {
        int j = jt * 16 + lm;
        #pragma unroll
        for (int r = 0; r < 8; ++r) {
          int b = brow + r + 8 * half;
          bf16_t hb = (bf16_t)fast_tanh(acc[jt][r]);
          hs[j * HS_LD + b] = hb;
          hout[(size_t)(t * B_ + b) * NH_ + j0g + j] = hb;
        }
      }
    }
    __syncthreads();

    // ---- Hebbian update: M[k,j] = lam*M[k,j] + eta * (h^T x)[j,k] ----------
    // D tile M-dim = j, N-dim = k  -> lane-contiguous RMW of Ms[j][k].
    {
      #pragma unroll
      for (int jt = 0; jt < 2; ++jt) {
        const bf16_t* hrow = &hs[(jt * 16 + lm) * HS_LD];
        v16bf a0 = cat8(*(const v8bf*)(hrow +      16 * half),
                        *(const v8bf*)(hrow +  8 + 16 * half));   // b = 0..31
        v16bf a1 = cat8(*(const v8bf*)(hrow + 32 + 16 * half),
                        *(const v8bf*)(hrow + 40 + 16 * half));   // b = 32..63
        for (int kt = wave; kt < NX_ / 16; kt += 4) {            // 8 tiles/wave
          int kn = kt * 16 + lm;
          const bf16_t* xcol = &xTs[kn * XT_LD];
          v16bf b0 = cat8(*(const v8bf*)(xcol +      16 * half),
                          *(const v8bf*)(xcol +  8 + 16 * half));
          v16bf b1 = cat8(*(const v8bf*)(xcol + 32 + 16 * half),
                          *(const v8bf*)(xcol + 40 + 16 * half));
          v8f d = vzero;
          d = __builtin_amdgcn_wmma_f32_16x16x32_bf16(
              false, a0, false, b0, (short)0, d, false, false);
          d = __builtin_amdgcn_wmma_f32_16x16x32_bf16(
              false, a1, false, b1, (short)0, d, false, false);
          #pragma unroll
          for (int r = 0; r < 8; ++r) {
            int jr = jt * 16 + r + 8 * half;
            float* mp = &Ms[jr * MS_LD + kn];
            *mp = lamV * (*mp) + etaV * d[r];
          }
        }
      }
    }
    __syncthreads();
  }
}

// ---------------------------------------------------------------------------
// Phase 2: y = tanh(h @ W^T + b), h bf16 [T*B, NH], W f32 [NY, NH].
// Each wave computes a 32x32 patch (2x2 WMMA tiles); block = 4 waves stacked
// in rows -> 128 rows x 32 cols. grid = (T*B/128, NY/32).
// ---------------------------------------------------------------------------
__global__ __launch_bounds__(128) void mpn_proj(
    const bf16_t* __restrict__ h,
    const float*  __restrict__ W,
    const float*  __restrict__ bias,
    float* __restrict__ y)
{
  const int tid  = threadIdx.x;
  const int wave = tid >> 5;
  const int lane = tid & 31;
  const int half = lane >> 4;
  const int lm   = lane & 15;
  const int m0 = blockIdx.x * 128 + wave * 32;
  const int n0 = blockIdx.y * 32;

  const v8f vzero = {0.f,0.f,0.f,0.f,0.f,0.f,0.f,0.f};
  v8f acc[2][2];
  acc[0][0] = vzero; acc[0][1] = vzero; acc[1][0] = vzero; acc[1][1] = vzero;

  for (int k0 = 0; k0 < NH_; k0 += 32) {
    v16bf a[2], bm[2];
    #pragma unroll
    for (int mt = 0; mt < 2; ++mt) {
      const bf16_t* hp = h + (size_t)(m0 + mt * 16 + lm) * NH_ + k0;
      a[mt] = cat8(*(const v8bf*)(hp + 16 * half),
                   *(const v8bf*)(hp + 8 + 16 * half));
    }
    #pragma unroll
    for (int nt = 0; nt < 2; ++nt) {
      const float4* wp =
          (const float4*)(W + (size_t)(n0 + nt * 16 + lm) * NH_ + k0 + 16 * half);
      v16bf bb;
      #pragma unroll
      for (int q = 0; q < 4; ++q) {
        float4 wv = wp[q];
        bb[4*q+0] = (bf16_t)wv.x; bb[4*q+1] = (bf16_t)wv.y;
        bb[4*q+2] = (bf16_t)wv.z; bb[4*q+3] = (bf16_t)wv.w;
      }
      bm[nt] = bb;
    }
    #pragma unroll
    for (int mt = 0; mt < 2; ++mt)
      #pragma unroll
      for (int nt = 0; nt < 2; ++nt)
        acc[mt][nt] = __builtin_amdgcn_wmma_f32_16x16x32_bf16(
            false, a[mt], false, bm[nt], (short)0, acc[mt][nt], false, false);
  }

  #pragma unroll
  for (int nt = 0; nt < 2; ++nt) {
    int n = n0 + nt * 16 + lm;
    float bv = bias[n];
    #pragma unroll
    for (int mt = 0; mt < 2; ++mt) {
      #pragma unroll
      for (int r = 0; r < 8; ++r) {
        int row = m0 + mt * 16 + r + 8 * half;
        y[(size_t)row * NY_ + n] = fast_tanh(acc[mt][nt][r] + bv);
      }
    }
  }
}

extern "C" void kernel_launch(void* const* d_in, const int* in_sizes, int n_in,
                              void* d_out, int out_size, void* d_ws, size_t ws_size,
                              hipStream_t stream) {
  const float* x   = (const float*)d_in[0];
  const float* Wi  = (const float*)d_in[1];
  const float* eta = (const float*)d_in[2];
  const float* lam = (const float*)d_in[3];
  const float* W   = (const float*)d_in[4];
  const float* b   = (const float*)d_in[5];
  bf16_t* hbuf = (bf16_t*)d_ws;           // T*B*NH bf16 = 64 MB scratch

  mpn_recur<<<NH_ / NC_, 128, 0, stream>>>(x, Wi, eta, lam, hbuf);
  mpn_proj<<<dim3((T_ * B_) / 128, NY_ / 32), 128, 0, stream>>>(hbuf, W, b,
                                                               (float*)d_out);
  (void)in_sizes; (void)n_in; (void)out_size; (void)ws_size;
}
